// ConditionalTransformer_3dda_81793357185192
// MI455X (gfx1250) — matrix operand order repaired and checked by reading, not compiled, not run on hardware
//
#include <hip/hip_runtime.h>
#include <hip/hip_bf16.h>
#include <cmath>

#define NB 16      // batch
#define DD 120     // model dim
#define NCTX 4096  // scene tokens (NOBS*N)
#define NQ 64      // trajectory tokens (T*NGP)
#define NG 8       // gripper tokens
#define NSAMP 819  // NCTX/5
#define NS 883     // NQ + NSAMP
#define SKPMAX 896 // padded score row stride for NS

typedef __attribute__((ext_vector_type(16))) _Float16 v16h;
typedef __attribute__((ext_vector_type(8)))  float    v8f;

// ---------------------------------------------------------------------------
// Batched GEMM via v_wmma_f32_16x16x32_f16.  One wave = 16x64 C strip
// (4 N-tiles reuse one A fragment -> 4 back-to-back WMMAs per k-step).
// Each k-step is 3 phases: (1) issue all raw loads in one clause,
// (2) convert into 4 distinct B fragments, (3) 4 WMMAs. Distinct registers
// per tile avoid the WMMA->VALU WAR hazard nops.
//   transB=1: C[b,m,n] = act(alpha*(sum_k A[b,m,k]*B[b,n,k] + bias[n]))
//   transB=0: C[b,m,n] = act(alpha*(sum_k A[b,m,k]*B[b,k,n] + bias[n]))
// Out-of-range M rows / N cols are address-clamped (their outputs are never
// stored); only the K tail uses select-to-zero.
// Fragment layouts used here (wave32; A/B K order contiguous, not the ISA table order):
//   A f16 16x32: lane L -> row L&15; pair v -> K = (L>>4)*16 + v*2
//   B f16 32x16: lane L -> col L&15; elem e -> K = (L>>4)*16 + e
//   C f32 16x16: lane L -> col L&15; vgpr r -> row (L>>4)*8 + r
// ---------------------------------------------------------------------------
__global__ __launch_bounds__(32) void k_gemm(
    const float* __restrict__ A, const float* __restrict__ Bm,
    const float* __restrict__ bias, float* __restrict__ C,
    int M, int N, int K, int lda, int ldb, int ldc,
    long long sA, long long sB, long long sC,
    float alpha, int relu, int transB)
{
  const int b = blockIdx.z;
  const float* __restrict__ Ab = A + (long long)b * sA;
  const float* __restrict__ Bb = Bm + (long long)b * sB;
  float* __restrict__ Cb = C + (long long)b * sC;
  const int lane = threadIdx.x & 31;
  const int hlf = lane >> 4, l16 = lane & 15;
  const int mrow = blockIdx.x * 16 + l16;
  const int mcl = (mrow < M) ? mrow : (M - 1);
  const float* __restrict__ Arow = Ab + (long long)mcl * lda;
  const float* __restrict__ AH = Arow + (hlf << 4);
  const int nbase = blockIdx.y * 64;

  v8f acc0 = {}, acc1 = {}, acc2 = {}, acc3 = {};
  int ncl[4];
  const float* __restrict__ BT[4];
#pragma unroll
  for (int t = 0; t < 4; ++t) {
    int n = nbase + t * 16 + l16;
    ncl[t] = (n < N) ? n : (N - 1);
    BT[t] = transB ? (Bb + (long long)ncl[t] * ldb + (hlf << 4))
                   : (Bb + ncl[t]);
  }

  const int kfull = K & ~31;
  if (transB) {
    for (int k0 = 0; k0 < kfull; k0 += 32) {
      // ---- phase 1: issue all loads (one clause) ----
      float2 ar[8], br0[8], br1[8], br2[8], br3[8];
#pragma unroll
      for (int v = 0; v < 8; ++v)
        ar[v] = *(const float2*)(AH + k0 + (v << 1));
#pragma unroll
      for (int q = 0; q < 8; ++q) br0[q] = *(const float2*)(BT[0] + k0 + 2 * q);
#pragma unroll
      for (int q = 0; q < 8; ++q) br1[q] = *(const float2*)(BT[1] + k0 + 2 * q);
#pragma unroll
      for (int q = 0; q < 8; ++q) br2[q] = *(const float2*)(BT[2] + k0 + 2 * q);
#pragma unroll
      for (int q = 0; q < 8; ++q) br3[q] = *(const float2*)(BT[3] + k0 + 2 * q);
      // ---- phase 2: convert into distinct fragments ----
      v16h af, bf0, bf1, bf2, bf3;
#pragma unroll
      for (int q = 0; q < 8; ++q) {
        af[2 * q] = (_Float16)ar[q].x;  af[2 * q + 1] = (_Float16)ar[q].y;
        bf0[2 * q] = (_Float16)br0[q].x; bf0[2 * q + 1] = (_Float16)br0[q].y;
        bf1[2 * q] = (_Float16)br1[q].x; bf1[2 * q + 1] = (_Float16)br1[q].y;
        bf2[2 * q] = (_Float16)br2[q].x; bf2[2 * q + 1] = (_Float16)br2[q].y;
        bf3[2 * q] = (_Float16)br3[q].x; bf3[2 * q + 1] = (_Float16)br3[q].y;
      }
      // ---- phase 3: 4 back-to-back WMMAs ----
      acc0 = __builtin_amdgcn_wmma_f32_16x16x32_f16(false, af, false, bf0, (short)0, acc0, false, false);
      acc1 = __builtin_amdgcn_wmma_f32_16x16x32_f16(false, af, false, bf1, (short)0, acc1, false, false);
      acc2 = __builtin_amdgcn_wmma_f32_16x16x32_f16(false, af, false, bf2, (short)0, acc2, false, false);
      acc3 = __builtin_amdgcn_wmma_f32_16x16x32_f16(false, af, false, bf3, (short)0, acc3, false, false);
    }
  } else {
    for (int k0 = 0; k0 < kfull; k0 += 32) {
      float2 ar[8];
      float br0[16], br1[16], br2[16], br3[16];
#pragma unroll
      for (int v = 0; v < 8; ++v)
        ar[v] = *(const float2*)(AH + k0 + (v << 1));
#pragma unroll
      for (int e = 0; e < 16; ++e) {
        long long k = k0 + (hlf << 4) + e;
        br0[e] = BT[0][k * ldb];
        br1[e] = BT[1][k * ldb];
        br2[e] = BT[2][k * ldb];
        br3[e] = BT[3][k * ldb];
      }
      v16h af, bf0, bf1, bf2, bf3;
#pragma unroll
      for (int q = 0; q < 8; ++q) { af[2 * q] = (_Float16)ar[q].x; af[2 * q + 1] = (_Float16)ar[q].y; }
#pragma unroll
      for (int e = 0; e < 16; ++e) {
        bf0[e] = (_Float16)br0[e]; bf1[e] = (_Float16)br1[e];
        bf2[e] = (_Float16)br2[e]; bf3[e] = (_Float16)br3[e];
      }
      acc0 = __builtin_amdgcn_wmma_f32_16x16x32_f16(false, af, false, bf0, (short)0, acc0, false, false);
      acc1 = __builtin_amdgcn_wmma_f32_16x16x32_f16(false, af, false, bf1, (short)0, acc1, false, false);
      acc2 = __builtin_amdgcn_wmma_f32_16x16x32_f16(false, af, false, bf2, (short)0, acc2, false, false);
      acc3 = __builtin_amdgcn_wmma_f32_16x16x32_f16(false, af, false, bf3, (short)0, acc3, false, false);
    }
  }

  if (kfull < K) {           // K tail: clamped loads + select-to-zero, no branches
    v16h af;
#pragma unroll
    for (int v = 0; v < 8; ++v) {
      int k = kfull + (hlf << 4) + (v << 1);
      float a0 = Arow[(k < K) ? k : 0];
      float a1 = Arow[(k + 1 < K) ? (k + 1) : 0];
      af[2 * v]     = (_Float16)((k     < K) ? a0 : 0.f);
      af[2 * v + 1] = (_Float16)((k + 1 < K) ? a1 : 0.f);
    }
#pragma unroll
    for (int t = 0; t < 4; ++t) {
      v16h bf;
#pragma unroll
      for (int e = 0; e < 16; ++e) {
        int k = kfull + (hlf << 4) + e;
        int kc = (k < K) ? k : 0;
        float bv = transB ? Bb[(long long)ncl[t] * ldb + kc]
                          : Bb[(long long)kc * ldb + ncl[t]];
        bf[e] = (_Float16)((k < K) ? bv : 0.f);
      }
      if (t == 0) acc0 = __builtin_amdgcn_wmma_f32_16x16x32_f16(false, af, false, bf, (short)0, acc0, false, false);
      if (t == 1) acc1 = __builtin_amdgcn_wmma_f32_16x16x32_f16(false, af, false, bf, (short)0, acc1, false, false);
      if (t == 2) acc2 = __builtin_amdgcn_wmma_f32_16x16x32_f16(false, af, false, bf, (short)0, acc2, false, false);
      if (t == 3) acc3 = __builtin_amdgcn_wmma_f32_16x16x32_f16(false, af, false, bf, (short)0, acc3, false, false);
    }
  }

#pragma unroll
  for (int t = 0; t < 4; ++t) {
    int n = nbase + t * 16 + l16;
    if (n < N) {
      float bb = bias ? bias[n] : 0.f;
      v8f acc = (t == 0) ? acc0 : (t == 1) ? acc1 : (t == 2) ? acc2 : acc3;
#pragma unroll
      for (int r = 0; r < 8; ++r) {
        int m = blockIdx.x * 16 + (hlf << 3) + r;
        if (m < M) {
          float v = (acc[r] + bb) * alpha;
          if (relu && v < 0.f) v = 0.f;
          Cb[(long long)m * ldc + n] = v;
        }
      }
    }
  }
}

// ---------------- softmax over len entries of a row with stride ld ---------
__global__ void k_softmax(float* __restrict__ S, int len, int ld)
{
  long long row = blockIdx.x;
  float* p = S + row * (long long)ld;
  __shared__ float sh[256];
  int tid = threadIdx.x;
  float m = -3.4e38f;
  for (int i = tid; i < len; i += 256) m = fmaxf(m, p[i]);
  sh[tid] = m; __syncthreads();
  for (int s = 128; s > 0; s >>= 1) { if (tid < s) sh[tid] = fmaxf(sh[tid], sh[tid + s]); __syncthreads(); }
  m = sh[0]; __syncthreads();
  float sum = 0.f;
  for (int i = tid; i < len; i += 256) { float e = expf(p[i] - m); p[i] = e; sum += e; }
  sh[tid] = sum; __syncthreads();
  for (int s = 128; s > 0; s >>= 1) { if (tid < s) sh[tid] += sh[tid + s]; __syncthreads(); }
  float inv = 1.f / sh[0];
  for (int i = tid; i < len; i += 256) p[i] *= inv;
}

// ---------------- y = LayerNorm(x + r)*g + b, one 128-thread block per row -
__global__ void k_add_ln(const float* __restrict__ x, const float* __restrict__ r,
                         const float* __restrict__ g, const float* __restrict__ bta,
                         float* __restrict__ y)
{
  long long row = blockIdx.x;
  int tid = threadIdx.x;
  __shared__ float sh[128];
  float v = 0.f;
  if (tid < DD) v = x[row * DD + tid] + r[row * DD + tid];
  sh[tid] = (tid < DD) ? v : 0.f; __syncthreads();
  for (int s = 64; s > 0; s >>= 1) { if (tid < s) sh[tid] += sh[tid + s]; __syncthreads(); }
  float mean = sh[0] / (float)DD; __syncthreads();
  float d = (tid < DD) ? (v - mean) : 0.f;
  sh[tid] = d * d; __syncthreads();
  for (int s = 64; s > 0; s >>= 1) { if (tid < s) sh[tid] += sh[tid + s]; __syncthreads(); }
  float var = sh[0] / (float)DD;
  if (tid < DD) y[row * DD + tid] = d * rsqrtf(var + 1e-5f) * g[tid] + bta[tid];
}

// ---------------- adaLN modulation: y = x*(1+scale[b,j]) + shift[b,j] ------
__global__ void k_adaln(const float* __restrict__ x, const float* __restrict__ mod,
                        float* __restrict__ y, int Sq, long long total)
{
  long long i = (long long)blockIdx.x * blockDim.x + threadIdx.x;
  if (i >= total) return;
  int j = (int)(i % DD);
  long long b = i / ((long long)Sq * DD);
  y[i] = x[i] * (1.f + mod[b * 240 + j]) + mod[b * 240 + 120 + j];
}

// ---------------- rotary (rope3d), in place; one thread per even/odd pair --
__global__ void k_rope(float* __restrict__ buf, const float* __restrict__ xyz,
                       long long total_pairs)
{
  long long i = (long long)blockIdx.x * blockDim.x + threadIdx.x;
  if (i >= total_pairs) return;
  long long tok = i / 60;
  int pj = (int)(i % 60) * 2;          // even channel index
  int c = pj / 40;                     // coordinate 0..2
  int f = (pj % 40) >> 1;              // frequency 0..19
  float ang = xyz[tok * 3 + c] * expf(-0.46051701859880916f * (float)f);
  float s = sinf(ang), co = cosf(ang);
  float x0 = buf[tok * DD + pj], x1 = buf[tok * DD + pj + 1];
  buf[tok * DD + pj]     = x0 * co - x1 * s;
  buf[tok * DD + pj + 1] = x1 * co + x0 * s;
}

// ---------------- timestep sin/cos embedding (B,120) -----------------------
__global__ void k_time_embed(const float* __restrict__ t, float* __restrict__ out)
{
  int i = blockIdx.x * blockDim.x + threadIdx.x;
  if (i >= NB * DD) return;
  int b = i / DD, j = i % DD;
  float tv = t[b];
  if (j < 60) out[i] = sinf(tv * expf(-0.1561074639317997f * (float)j));
  else        out[i] = cosf(tv * expf(-0.1561074639317997f * (float)(j - 60)));
}

__global__ void k_add2(const float* a, const float* b, float* c, int n)
{ int i = blockIdx.x * blockDim.x + threadIdx.x; if (i < n) c[i] = a[i] + b[i]; }

__global__ void k_silu(const float* a, float* y, int n)
{ int i = blockIdx.x * blockDim.x + threadIdx.x; if (i < n) { float x = a[i]; y[i] = x / (1.f + expf(-x)); } }

__global__ void k_bcast(const float* ge, float* gf, int n)
{ int i = blockIdx.x * blockDim.x + threadIdx.x; if (i < n) gf[i] = ge[i % (NG * DD)]; }

// ---------------- farthest point sampling over 120-d features --------------
__global__ __launch_bounds__(512) void k_fps(const float* __restrict__ sf, int* __restrict__ idx)
{
  int b = blockIdx.x, tid = threadIdx.x;
  __shared__ float  dist[NCTX];
  __shared__ float4 lastf[30];
  __shared__ float  redv[512];
  __shared__ int    redi[512];
  const float* X = sf + (long long)b * NCTX * DD;
  for (int i = tid; i < NCTX; i += 512) dist[i] = 3.4e38f;
  if (tid == 0) idx[b * NSAMP] = 0;
  int last = 0;
  __syncthreads();
  for (int step = 1; step < NSAMP; ++step) {
    const float4* lp = (const float4*)(X + (long long)last * DD);
    for (int j = tid; j < 30; j += 512) lastf[j] = lp[j];
    __syncthreads();
    float bestv = -1.f; int besti = 0;
    for (int i = tid; i < NCTX; i += 512) {
      const float4* xr = (const float4*)(X + (long long)i * DD);
      float d = 0.f;
#pragma unroll 6
      for (int j = 0; j < 30; ++j) {
        float4 a = xr[j], l = lastf[j];
        float dx = a.x - l.x, dy = a.y - l.y, dz = a.z - l.z, dw = a.w - l.w;
        d += dx * dx + dy * dy + dz * dz + dw * dw;
      }
      float nd = fminf(dist[i], d);
      dist[i] = nd;
      if (nd > bestv) { bestv = nd; besti = i; }   // strict > keeps first index
    }
    redv[tid] = bestv; redi[tid] = besti; __syncthreads();
    for (int s = 256; s > 0; s >>= 1) {
      if (tid < s) {
        float ov = redv[tid + s]; int oi = redi[tid + s];
        if (ov > redv[tid] || (ov == redv[tid] && oi < redi[tid])) { redv[tid] = ov; redi[tid] = oi; }
      }
      __syncthreads();
    }
    last = redi[0];
    if (tid == 0) idx[b * NSAMP + step] = last;
    __syncthreads();
  }
}

// ---------------- build feats (B,883,120) and pos_xyz (B,883,3) ------------
__global__ void k_gather(const float* __restrict__ gq, const float* __restrict__ sf,
                         const int* __restrict__ idx, const float* __restrict__ sample,
                         const float* __restrict__ sp, float* __restrict__ feats,
                         float* __restrict__ pos_xyz)
{
  long long i = (long long)blockIdx.x * blockDim.x + threadIdx.x;
  long long total = (long long)NB * NS * DD;
  if (i >= total) return;
  int j = (int)(i % DD);
  long long bs = i / DD;
  int s = (int)(bs % NS);
  int b = (int)(bs / NS);
  float f;
  if (s < NQ) {
    f = gq[((long long)b * NQ + s) * DD + j];
    if (j < 3) pos_xyz[((long long)b * NS + s) * 3 + j] = sample[((long long)b * NQ + s) * 3 + j];
  } else {
    int src = idx[b * NSAMP + (s - NQ)];
    f = sf[((long long)b * NCTX + src) * DD + j];
    if (j < 3) pos_xyz[((long long)b * NS + s) * 3 + j] = sp[((long long)b * NCTX + src) * 3 + j];
  }
  feats[i] = f;
}

// ===========================================================================
// Host side
// ===========================================================================
struct Lin { const float* w; const float* b; };
struct Layer {
  Lin adaln, ffw1, ffw2, ffw_adaln;
  const float *ffw_ln_b, *ffw_ln_g, *in_b, *in_w, *ln_b, *ln_g;
  Lin out; int has_adaln;
};
struct Ws {
  float *sf, *gfeat, *qfeat, *feats, *pos_xyz;
  float *aq, *qb, *kb, *vb, *att, *ffw, *proj, *scores, *mod;
  float *te0, *t1, *t2, *g1, *g2, *diff, *silu;
  float *pf, *pf2;
  int *idx;
};

static inline void gemm(hipStream_t st, const float* A, const float* Bm,
                        const float* bias, float* C,
                        int M, int N, int K, int lda, int ldb, int ldc,
                        long long sA, long long sB, long long sC, int batch,
                        float alpha, int relu, int transB)
{
  dim3 grid((M + 15) / 16, (N + 63) / 64, batch);
  k_gemm<<<grid, 32, 0, st>>>(A, Bm, bias, C, M, N, K, lda, ldb, ldc,
                              sA, sB, sC, alpha, relu, transB);
}

static void attn_block(hipStream_t st, const Layer& L, float* q, int Sq,
                       const float* kv, int Sk, int nheads,
                       const float* silu_ts, const float* q_xyz,
                       const float* k_xyz, Ws& W)
{
  const int dh = DD / nheads;
  const int Skp = (Sk + 31) & ~31;       // padded score row stride (vector aligned)
  const float scale = 1.0f / sqrtf((float)dh);
  const long long Mq = (long long)NB * Sq;
  const long long Mk = (long long)NB * Sk;
  const float* aq = q;
  if (L.has_adaln && silu_ts) {
    gemm(st, silu_ts, L.adaln.w, L.adaln.b, W.mod, NB, 240, DD, DD, DD, 240, 0, 0, 0, 1, 1.f, 0, 1);
    long long tot = Mq * DD;
    k_adaln<<<(unsigned)((tot + 255) / 256), 256, 0, st>>>(q, W.mod, W.aq, Sq, tot);
    aq = W.aq;
  }
  const float* kvin = kv ? kv : aq;
  gemm(st, aq,   L.in_w,               L.in_b,          W.qb, (int)Mq, DD, DD, DD, DD, DD, 0, 0, 0, 1, scale, 0, 1);
  gemm(st, kvin, L.in_w + DD * DD,     L.in_b + DD,     W.kb, (int)Mk, DD, DD, DD, DD, DD, 0, 0, 0, 1, 1.f,   0, 1);
  gemm(st, kvin, L.in_w + 2 * DD * DD, L.in_b + 2 * DD, W.vb, (int)Mk, DD, DD, DD, DD, DD, 0, 0, 0, 1, 1.f,   0, 1);
  if (q_xyz) {
    long long pq = Mq * 60, pk = Mk * 60;
    k_rope<<<(unsigned)((pq + 255) / 256), 256, 0, st>>>(W.qb, q_xyz, pq);
    k_rope<<<(unsigned)((pk + 255) / 256), 256, 0, st>>>(W.kb, k_xyz, pk);
  }
  for (int h = 0; h < nheads; ++h) {
    gemm(st, W.qb + h * dh, W.kb + h * dh, nullptr, W.scores,
         Sq, Sk, dh, DD, DD, Skp,
         (long long)Sq * DD, (long long)Sk * DD, (long long)Sq * Skp, NB, 1.f, 0, 1);
    k_softmax<<<(unsigned)(NB * Sq), 256, 0, st>>>(W.scores, Sk, Skp);
    gemm(st, W.scores, W.vb + h * dh, nullptr, W.att + h * dh,
         Sq, dh, Sk, Skp, DD, DD,
         (long long)Sq * Skp, (long long)Sk * DD, (long long)Sq * DD, NB, 1.f, 0, 0);
  }
  gemm(st, W.att, L.out.w, L.out.b, W.proj, (int)Mq, DD, DD, DD, DD, DD, 0, 0, 0, 1, 1.f, 0, 1);
  k_add_ln<<<(unsigned)Mq, 128, 0, st>>>(q, W.proj, L.ln_g, L.ln_b, q);
  const float* ax = q;
  if (L.has_adaln && silu_ts) {
    gemm(st, silu_ts, L.ffw_adaln.w, L.ffw_adaln.b, W.mod, NB, 240, DD, DD, DD, 240, 0, 0, 0, 1, 1.f, 0, 1);
    long long tot = Mq * DD;
    k_adaln<<<(unsigned)((tot + 255) / 256), 256, 0, st>>>(q, W.mod, W.aq, Sq, tot);
    ax = W.aq;
  }
  gemm(st, ax,    L.ffw1.w, L.ffw1.b, W.ffw,  (int)Mq, DD, DD, DD, DD, DD, 0, 0, 0, 1, 1.f, 1, 1);
  gemm(st, W.ffw, L.ffw2.w, L.ffw2.b, W.proj, (int)Mq, DD, DD, DD, DD, DD, 0, 0, 0, 1, 1.f, 0, 1);
  k_add_ln<<<(unsigned)Mq, 128, 0, st>>>(q, W.proj, L.ffw_ln_g, L.ffw_ln_b, q);
}

extern "C" void kernel_launch(void* const* d_in, const int* in_sizes, int n_in,
                              void* d_out, int out_size, void* d_ws, size_t ws_size,
                              hipStream_t stream)
{
  (void)in_sizes; (void)out_size; (void)ws_size;
  // -------- parse params (jax tree_leaves order: dict keys sorted) ---------
  const bool blob = (n_in <= 16);
  const float* prm = blob ? (const float*)d_in[0] : nullptr;
  int li = 0; size_t off = 0;
  auto leaf = [&](size_t n) -> const float* {
    if (blob) { const float* p = prm + off; off += n; return p; }
    return (const float*)d_in[li++];
  };
  auto lin = [&](int din, int dout) {
    Lin l; l.b = leaf((size_t)dout); l.w = leaf((size_t)dout * din); return l;
  };
  auto layer = [&](int adaln) {
    Layer L{}; L.has_adaln = adaln;
    if (adaln) L.adaln = lin(DD, 240);
    L.ffw1 = lin(DD, DD);
    L.ffw2 = lin(DD, DD);
    if (adaln) L.ffw_adaln = lin(DD, 240);
    L.ffw_ln_b = leaf(DD); L.ffw_ln_g = leaf(DD);
    L.in_b = leaf(3 * DD); L.in_w = leaf((size_t)3 * DD * DD);
    L.ln_b = leaf(DD); L.ln_g = leaf(DD);
    L.out = lin(DD, DD);
    return L;
  };
  Lin cg1 = lin(960, DD);
  Lin cg2 = lin(DD, DD);
  Layer cross[2]; for (int i = 0; i < 2; ++i) cross[i] = layer(1);
  Lin ctx = lin(60, DD);
  const float* grip_embed = leaf((size_t)NG * DD);
  Layer grip[3]; for (int i = 0; i < 3; ++i) grip[i] = layer(0);
  Lin pos_proj = lin(DD, DD);
  Layer possa[2]; for (int i = 0; i < 2; ++i) possa[i] = layer(1);
  Lin pp1 = lin(DD, DD);
  Lin pp2 = lin(DD, 3);
  Layer selfa[4]; for (int i = 0; i < 4; ++i) selfa[i] = layer(1);
  Lin time1 = lin(DD, DD);
  Lin time2 = lin(DD, DD);
  Lin traj = lin(3, DD);

  const int ti = blob ? 1 : li;
  const float* sample = (const float*)d_in[ti + 0];   // (B,64,3) flat
  const float* tstep  = (const float*)d_in[ti + 1];   // (B,)
  const float* og     = (const float*)d_in[ti + 3];   // (B,8,3)
  const float* sp     = (const float*)d_in[ti + 4];   // (B,4096,3)
  const float* scf    = (const float*)d_in[ti + 5];   // (B,4096,60)
  float* out = (float*)d_out;

  // -------- workspace carve-up --------------------------------------------
  char* base = (char*)d_ws; size_t o = 0;
  auto alloc_f = [&](size_t n) { o = (o + 255) & ~(size_t)255; float* p = (float*)(base + o); o += n * 4; return p; };
  auto alloc_i = [&](size_t n) { o = (o + 255) & ~(size_t)255; int*   p = (int*)(base + o);   o += n * 4; return p; };
  Ws W{};
  W.sf      = alloc_f((size_t)NB * NCTX * DD);
  W.gfeat   = alloc_f((size_t)NB * NG * DD);
  W.qfeat   = alloc_f((size_t)NB * NQ * DD);
  W.feats   = alloc_f((size_t)NB * NS * DD);
  W.pos_xyz = alloc_f((size_t)NB * NS * 3);
  W.aq      = alloc_f((size_t)NB * NS * DD);
  W.qb      = alloc_f((size_t)NB * NS * DD);
  W.kb      = alloc_f((size_t)NB * NCTX * DD);
  W.vb      = alloc_f((size_t)NB * NCTX * DD);
  W.att     = alloc_f((size_t)NB * NS * DD);
  W.ffw     = alloc_f((size_t)NB * NS * DD);
  W.proj    = alloc_f((size_t)NB * NS * DD);
  W.scores  = alloc_f((size_t)NB * NS * SKPMAX);  // padded; covers all attn shapes
  W.mod     = alloc_f((size_t)NB * 240);
  W.te0 = alloc_f(NB * DD); W.t1 = alloc_f(NB * DD); W.t2 = alloc_f(NB * DD);
  W.g1  = alloc_f(NB * DD); W.g2 = alloc_f(NB * DD);
  W.diff = alloc_f(NB * DD); W.silu = alloc_f(NB * DD);
  W.pf  = alloc_f((size_t)NB * NQ * DD);
  W.pf2 = alloc_f((size_t)NB * NQ * DD);
  W.idx = alloc_i((size_t)NB * NSAMP);

  // -------- 1) scene features: sf = relu(scf @ ctx.w^T + ctx.b) -----------
  gemm(stream, scf, ctx.w, ctx.b, W.sf, NB * NCTX, DD, 60, 60, 60, DD, 0, 0, 0, 1, 1.f, 1, 1);

  // -------- 2) gripper token stack (3 cross layers, 3 heads, no adaln) ----
  { int n = NB * NG * DD; k_bcast<<<(n + 255) / 256, 256, 0, stream>>>(grip_embed, W.gfeat, n); }
  for (int i = 0; i < 3; ++i)
    attn_block(stream, grip[i], W.gfeat, NG, W.sf, NCTX, 3, nullptr, og, sp, W);

  // -------- 3) conditioning: diff_ts = timeMLP(t) + gripMLP(gfeat) --------
  k_time_embed<<<(NB * DD + 255) / 256, 256, 0, stream>>>(tstep, W.te0);
  gemm(stream, W.te0, time1.w, time1.b, W.t1, NB, DD, DD, DD, DD, DD, 0, 0, 0, 1, 1.f, 1, 1);
  gemm(stream, W.t1,  time2.w, time2.b, W.t2, NB, DD, DD, DD, DD, DD, 0, 0, 0, 1, 1.f, 0, 1);
  gemm(stream, W.gfeat, cg1.w, cg1.b, W.g1, NB, DD, 960, 960, 960, DD, 0, 0, 0, 1, 1.f, 1, 1);
  gemm(stream, W.g1,    cg2.w, cg2.b, W.g2, NB, DD, DD, DD, DD, DD, 0, 0, 0, 1, 1.f, 0, 1);
  { int n = NB * DD;
    k_add2<<<(n + 255) / 256, 256, 0, stream>>>(W.t2, W.g2, W.diff, n);
    k_silu<<<(n + 255) / 256, 256, 0, stream>>>(W.diff, W.silu, n); }

  // -------- 4) trajectory tokens + 2 cross-attn layers --------------------
  gemm(stream, sample, traj.w, traj.b, W.qfeat, NB * NQ, DD, 3, 3, 3, DD, 0, 0, 0, 1, 1.f, 0, 1);
  for (int i = 0; i < 2; ++i)
    attn_block(stream, cross[i], W.qfeat, NQ, W.sf, NCTX, 4, W.silu, sample, sp, W);

  // -------- 5) farthest point sampling + token gather ---------------------
  k_fps<<<NB, 512, 0, stream>>>(W.sf, W.idx);
  { long long tot = (long long)NB * NS * DD;
    k_gather<<<(unsigned)((tot + 255) / 256), 256, 0, stream>>>(
        W.qfeat, W.sf, W.idx, sample, sp, W.feats, W.pos_xyz); }

  // -------- 6) self-attention stacks (4 + 2 layers) -----------------------
  for (int i = 0; i < 4; ++i)
    attn_block(stream, selfa[i], W.feats, NS, nullptr, NS, 4, W.silu, W.pos_xyz, W.pos_xyz, W);
  for (int i = 0; i < 2; ++i)
    attn_block(stream, possa[i], W.feats, NS, nullptr, NS, 4, W.silu, W.pos_xyz, W.pos_xyz, W);

  // -------- 7) output heads on first 64 tokens ----------------------------
  gemm(stream, W.feats, pos_proj.w, pos_proj.b, W.pf, NQ, DD, DD, DD, DD, DD,
       (long long)NS * DD, 0, (long long)NQ * DD, NB, 1.f, 0, 1);
  gemm(stream, W.pf,  pp1.w, pp1.b, W.pf2, NB * NQ, DD, DD, DD, DD, DD, 0, 0, 0, 1, 1.f, 1, 1);
  gemm(stream, W.pf2, pp2.w, pp2.b, out,   NB * NQ, 3,  DD, DD, DD, 3,  0, 0, 0, 1, 1.f, 0, 1);
}
